// GCN_15925738733667
// MI455X (gfx1250) — matrix operand order repaired and checked by reading, not compiled, not run on hardware
//
#include <hip/hip_runtime.h>

typedef __bf16 v16bf __attribute__((ext_vector_type(16)));
typedef __bf16 v8bf  __attribute__((ext_vector_type(8)));
typedef float  v8f   __attribute__((ext_vector_type(8)));

// ---------------- degree / normalization ----------------

__global__ void gcn_init_deg(float* __restrict__ deg, int N) {
    int i = blockIdx.x * blockDim.x + threadIdx.x;
    if (i < N) deg[i] = 1.0f;  // self-loop contributes 1 to every node's degree
}

__global__ void gcn_accum_deg(const long long* __restrict__ dst, float* __restrict__ deg, int E) {
    int e = blockIdx.x * blockDim.x + threadIdx.x;
    if (e < E) atomicAdd(&deg[(int)dst[e]], 1.0f);
}

__global__ void gcn_rsqrt(float* __restrict__ deg, int N) {
    int i = blockIdx.x * blockDim.x + threadIdx.x;
    if (i < N) deg[i] = rsqrtf(deg[i]);   // deg >= 1 always (self-loops)
}

// norm[e] = dinv[src]*dinv[dst] for edges, norm[E+i] = dinv[i]^2 for self-loops
__global__ void gcn_edge_norm(const long long* __restrict__ src,
                              const long long* __restrict__ dst,
                              const float* __restrict__ dinv,
                              float* __restrict__ norm, int E, int N) {
    long i = (long)blockIdx.x * blockDim.x + threadIdx.x;
    if (i < E) {
        norm[i] = dinv[(int)src[i]] * dinv[(int)dst[i]];
    } else if (i < (long)E + N) {
        float di = dinv[i - E];
        norm[i] = di * di;
    }
}

// ---------------- dense transform: H = relu(X + bias_prev) @ W via v_wmma_f32_16x16x32_bf16 ----
// Block: 16 output rows x FOUT output cols; one wave per 16-col tile.
// K = 128 -> four k-steps of 32. A tile and W (transposed) staged in LDS as bf16 with padded
// stride (136 = 17*8) so fragment ds_load_b128 lanes start 4 banks apart (conflict-free) while
// keeping 16-byte alignment. Optional bias+ReLU of the *previous* layer is fused into staging.

template <int FOUT, bool FUSE_PREV>
__global__ __launch_bounds__(FOUT * 2) void gcn_gemm_wmma(
    const float* __restrict__ X, const float* __restrict__ W,
    const float* __restrict__ prevBias, float* __restrict__ H, int N) {
    constexpr int K  = 128;
    constexpr int KP = 136;                         // padded LDS row stride (multiple of 8)
    __shared__ alignas(16) __bf16 As[16 * KP];      // row-major 16 x K (padded)
    __shared__ alignas(16) __bf16 Wt[FOUT * KP];    // transposed: Wt[col][k] (padded)

    const int tid      = threadIdx.x;
    const int nthreads = FOUT * 2;
    const int rowBase  = blockIdx.x * 16;

    // cooperative A-tile load, fused relu(x + b_prev), f32->bf16 convert
    for (int i = tid; i < 16 * K; i += nthreads) {
        int r = i >> 7, k = i & (K - 1);
        int row = rowBase + r;
        float v = (row < N) ? X[(long)row * K + k] : 0.0f;
        if (FUSE_PREV) v = fmaxf(v + prevBias[k], 0.0f);
        As[r * KP + k] = (__bf16)v;
    }
    // cooperative W load, transposed into LDS so B-fragments are contiguous
    for (int i = tid; i < FOUT * K; i += nthreads) {
        int k = i / FOUT, col = i % FOUT;
        Wt[col * KP + k] = (__bf16)W[i];
    }
    __syncthreads();

    const int wave    = tid >> 5;
    const int lane    = tid & 31;
    const int half    = lane >> 4;   // 0: lanes 0-15, 1: lanes 16-31
    const int l16     = lane & 15;
    const int colTile = wave * 16;

    v8f c = {0.f, 0.f, 0.f, 0.f, 0.f, 0.f, 0.f, 0.f};

#pragma unroll
    for (int ks = 0; ks < K; ks += 32) {
        // A 16x32 bf16 layout: lane holds row l16, K in {half*8..+7} U {16+half*8..+7}
        v8bf alo = *(const v8bf*)&As[l16 * KP + ks + half * 8];
        v8bf ahi = *(const v8bf*)&As[l16 * KP + ks + 16 + half * 8];
        // B 32x16 bf16 layout: lane holds col l16, K in {half*8..+7} U {16+half*8..+7}
        v8bf blo = *(const v8bf*)&Wt[(colTile + l16) * KP + ks + half * 8];
        v8bf bhi = *(const v8bf*)&Wt[(colTile + l16) * KP + ks + 16 + half * 8];
        v16bf a, b;
#pragma unroll
        for (int i = 0; i < 8; ++i) {
            a[i] = alo[i]; a[i + 8] = ahi[i];
            b[i] = blo[i]; b[i + 8] = bhi[i];
        }
        c = __builtin_amdgcn_wmma_f32_16x16x32_bf16(
                /*neg_a=*/false, a, /*neg_b=*/false, b,
                /*c_mod=*/(short)0, c, /*reuse_a=*/false, /*reuse_b=*/false);
    }

    // D layout: VGPR r -> M = r + 8*half, N = l16
    const int row0 = rowBase + half * 8;
    float* outp = H + (long)row0 * FOUT + colTile + l16;
    if (rowBase + 16 <= N) {       // uniform fast path: straight-line stores
#pragma unroll
        for (int r = 0; r < 8; ++r) outp[(long)r * FOUT] = c[r];
    } else {
#pragma unroll
        for (int r = 0; r < 8; ++r)
            if (row0 + r < N) outp[(long)r * FOUT] = c[r];
    }
}

// ---------------- message aggregation: out[dst] += norm * h[src] ----------------
// One wave per message; entries [E, E+N) are the self-loops. Requires `out` pre-zeroed.

template <int F>
__global__ void gcn_aggregate(const float* __restrict__ H,
                              const long long* __restrict__ src,
                              const long long* __restrict__ dst,
                              const float* __restrict__ norm,
                              float* __restrict__ out, int E, int N) {
    const int warpsPerBlock = blockDim.x >> 5;
    const long gid  = (long)blockIdx.x * warpsPerBlock + (threadIdx.x >> 5);
    const int  lane = threadIdx.x & 31;
    const long total = (long)E + N;
    if (gid >= total) return;

    long s, d;
    if (gid < E) {
        s = (long)src[gid];
        d = (long)dst[gid];
    } else {
        s = d = gid - E;
    }
    const float nrm = norm[gid];
    const float* hs = H + s * F;
    float*       od = out + d * F;
    if (F == 128) {
        float4 v = ((const float4*)hs)[lane];
        float* p = od + lane * 4;
        atomicAdd(p + 0, v.x * nrm);
        atomicAdd(p + 1, v.y * nrm);
        atomicAdd(p + 2, v.z * nrm);
        atomicAdd(p + 3, v.w * nrm);
    } else {  // F == 64
        float2 v = ((const float2*)hs)[lane];
        float* p = od + lane * 2;
        atomicAdd(p + 0, v.x * nrm);
        atomicAdd(p + 1, v.y * nrm);
    }
}

// ---------------- final bias (no activation), in place; F must be a power of two ----------------

__global__ void gcn_bias(float* __restrict__ out, const float* __restrict__ bias,
                         long total, int Fmask) {
    long i = (long)blockIdx.x * blockDim.x + threadIdx.x;
    if (i >= total) return;
    out[i] += bias[(int)i & Fmask];
}

// ---------------- launcher ----------------

extern "C" void kernel_launch(void* const* d_in, const int* in_sizes, int n_in,
                              void* d_out, int out_size, void* d_ws, size_t ws_size,
                              hipStream_t stream) {
    const float*     x  = (const float*)d_in[0];
    const long long* ei = (const long long*)d_in[1];
    const float*     W0 = (const float*)d_in[2];
    const float*     b0 = (const float*)d_in[3];
    const float*     W1 = (const float*)d_in[4];
    const float*     b1 = (const float*)d_in[5];
    const float*     W2 = (const float*)d_in[6];
    const float*     b2 = (const float*)d_in[7];

    const int N = in_sizes[0] / 128;   // 100000
    const int E = in_sizes[1] / 2;     // 1600000
    const long long* srcp = ei;
    const long long* dstp = ei + E;

    // workspace: dinv [N] | norm [E+N] | h [N*128] | agg [N*128]
    char*  ws    = (char*)d_ws;
    size_t offD  = ((size_t)N * sizeof(float) + 255) & ~(size_t)255;
    size_t offNm = offD + ((((size_t)E + N) * sizeof(float) + 255) & ~(size_t)255);
    float* dinv  = (float*)ws;
    float* norm  = (float*)(ws + offD);
    float* hbuf  = (float*)(ws + offNm);
    float* agg   = (float*)(ws + offNm + (size_t)N * 128 * sizeof(float));

    const int  T        = 256;
    const int  nBlocksN = (N + T - 1) / T;
    const int  nTiles   = (N + 15) / 16;
    const long totMsg   = (long)E + N;
    const int  aggBlk   = (int)((totMsg + 7) / 8);   // 8 waves / block

    // normalization coefficients (shared by all three layers)
    gcn_init_deg<<<nBlocksN, T, 0, stream>>>(dinv, N);
    gcn_accum_deg<<<(E + T - 1) / T, T, 0, stream>>>(dstp, dinv, E);
    gcn_rsqrt<<<nBlocksN, T, 0, stream>>>(dinv, N);
    gcn_edge_norm<<<(int)((totMsg + T - 1) / T), T, 0, stream>>>(srcp, dstp, dinv, norm, E, N);

    // ---- layer 0: h = x @ W0 ; agg = A_hat h ----
    gcn_gemm_wmma<128, false><<<nTiles, 256, 0, stream>>>(x, W0, nullptr, hbuf, N);
    hipMemsetAsync(agg, 0, (size_t)N * 128 * sizeof(float), stream);
    gcn_aggregate<128><<<aggBlk, 256, 0, stream>>>(hbuf, srcp, dstp, norm, agg, E, N);

    // ---- layer 1: h = relu(agg + b0) @ W1 ; agg = A_hat h  (bias+relu fused into staging) ----
    gcn_gemm_wmma<128, true><<<nTiles, 256, 0, stream>>>(agg, W1, b0, hbuf, N);
    hipMemsetAsync(agg, 0, (size_t)N * 128 * sizeof(float), stream);
    gcn_aggregate<128><<<aggBlk, 256, 0, stream>>>(hbuf, srcp, dstp, norm, agg, E, N);

    // ---- layer 2: h = relu(agg + b1) @ W2 ; d_out = A_hat h + b2 ----
    gcn_gemm_wmma<64, true><<<nTiles, 128, 0, stream>>>(agg, W2, b1, hbuf, N);
    hipMemsetAsync(d_out, 0, (size_t)out_size * sizeof(float), stream);
    gcn_aggregate<64><<<aggBlk, 256, 0, stream>>>(hbuf, srcp, dstp, norm, (float*)d_out, E, N);
    gcn_bias<<<(int)(((long)N * 64 + T - 1) / T), T, 0, stream>>>((float*)d_out, b2, (long)N * 64, 63);
}
